// TransformerDecoderV3_39213051412918
// MI455X (gfx1250) — matrix operand order repaired and checked by reading, not compiled, not run on hardware
//
#include <hip/hip_runtime.h>
#include <cmath>
#include <cstdint>

// ---------------------------------------------------------------------------
// Model constants (from the reference)
// ---------------------------------------------------------------------------
#define BB     2
#define TT     768
#define MTOK   (BB*TT)        // 1536 tokens
#define HIDD   1024
#define VOC    2048
#define NCODE  8
#define FFD    2816
#define RMSEPS 1e-5f

// ---------------------------------------------------------------------------
// WMMA types (CDNA5 gfx1250, wave32)
// ---------------------------------------------------------------------------
typedef __attribute__((ext_vector_type(8)))  float  v8f;
typedef __attribute__((ext_vector_type(16))) __bf16 bfx16;
typedef __attribute__((ext_vector_type(8)))  __bf16 bfx8;

union V16U { bfx16 v; bfx8 h[2]; };

// ---------------------------------------------------------------------------
// Batched GEMM: C[M,N] = A[M,K] * B (+bias)(+res), bf16 WMMA, f32 accumulate.
//   BSRC=0: B is f32 [K,N] row-major  (convert + transpose while staging)
//   BSRC=1: B is f32 [N,K] row-major  (computes A*B^T; convert while staging)
//   BSRC=2: B is bf16 [N,K] row-major (GLOBAL_LOAD_ASYNC_TO_LDS_B128 direct copy)
// Batch: z -> (zo = z/bH, zi = z%bH); A += zo*sAo + zi*sAi;
//        B += zo*sBo + (zi/nrep)*sBi;  C += zo*sCo + zi*sCi.
//
// Tiles: block = 256 threads = 8 wave32 waves arranged 4 (rows) x 2 (cols).
//   BM = 64*MF, BN = 32*NF, BK = 32; each wave: MF x NF v_wmma per K-step.
// Double-buffered LDS, one barrier per K-step, global_prefetch of K+2 tiles.
// ---------------------------------------------------------------------------
#define BKK 32
#define LDP 40   // padded LDS row stride in bf16 elems (80B -> conflict-free frags)

template <int BSRC, int MF, int NF>
__global__ __launch_bounds__(256)
void gemm_bf16_wmma(const float* __restrict__ A, long long lda,
                    const float* __restrict__ B, long long ldb,
                    float* C, long long ldc,
                    const float* __restrict__ bias,
                    const float* res,
                    int M, int N, int K,
                    int bH, int nrep,
                    long long sAo, long long sAi,
                    long long sBo, long long sBi,
                    long long sCo, long long sCi)
{
    constexpr int BMt = 64 * MF;          // 64 or 128
    constexpr int BNt = 32 * NF;          // 64 or 128
    constexpr int NSH = (BNt == 128) ? 5 : 4;   // log2(BNt/4)

    __shared__ alignas(16) __bf16 sA[2][BMt * LDP];
    __shared__ alignas(16) __bf16 sB[2][BNt * LDP];

    const int z  = blockIdx.z;
    const int zo = z / bH;
    const int zi = z - zo * bH;
    A += zo * sAo + (long long)zi * sAi;
    C += zo * sCo + (long long)zi * sCi;
    if (res) res += zo * sCo + (long long)zi * sCi;
    const long long bOff = zo * sBo + (long long)(zi / nrep) * sBi;
    const float*  Bf  = B + bOff;                        // BSRC 0/1
    const __bf16* Bh  = (const __bf16*)B + bOff;         // BSRC 2

    const int m0   = blockIdx.y * BMt;
    const int n0   = blockIdx.x * BNt;
    const int tid  = threadIdx.x;
    const int lane = tid & 31;
    const int wid  = tid >> 5;
    const int wm   = (wid & 3) * 16 * MF;   // wave row offset inside tile
    const int wn   = (wid >> 2) * 16 * NF;  // wave col offset inside tile
    const int l16  = lane & 15;
    const int hi   = lane >> 4;

    v8f acc[MF][NF];
    #pragma unroll
    for (int m = 0; m < MF; ++m)
        #pragma unroll
        for (int n = 0; n < NF; ++n) acc[m][n] = (v8f){};

    // ---- staging helpers ----
    auto stageA = [&](int buf, int k0) {
        #pragma unroll
        for (int i = 0; i < BMt / 32; ++i) {
            int pos = tid + i * 256;           // (BMt*8) float4 slots
            int r   = pos >> 3;
            int cc  = (pos & 7) << 2;
            float4 f = *(const float4*)(A + (long long)(m0 + r) * lda + (k0 + cc));
            __bf16* d = &sA[buf][r * LDP + cc];
            d[0] = (__bf16)f.x; d[1] = (__bf16)f.y; d[2] = (__bf16)f.z; d[3] = (__bf16)f.w;
        }
    };
    auto stageB = [&](int buf, int k0) {
        if (BSRC == 2) {
            // bf16 [N,K]: per-lane 16B async copy, global -> LDS, ASYNCcnt-tracked
            #pragma unroll
            for (int i = 0; i < BNt / 64; ++i) {
                int pos = tid + i * 256;       // BNt*4 16-byte chunks
                int r   = pos >> 2;            // n: 0..BNt-1
                int cc  = (pos & 3) << 3;      // k: 0,8,16,24
                const __bf16* gp = Bh + (long long)(n0 + r) * ldb + (k0 + cc);
                unsigned ldst = (unsigned)(uintptr_t)(&sB[buf][r * LDP + cc]);
                asm volatile("global_load_async_to_lds_b128 %0, %1, off"
                             :: "v"(ldst), "v"((unsigned long long)(uintptr_t)gp)
                             : "memory");
            }
        } else {
            #pragma unroll
            for (int i = 0; i < BNt / 32; ++i) {
                int pos = tid + i * 256;
                if (BSRC == 1) {                    // f32 [N,K]: direct copy
                    int r  = pos >> 3;              // n
                    int cc = (pos & 7) << 2;        // k
                    float4 f = *(const float4*)(Bf + (long long)(n0 + r) * ldb + (k0 + cc));
                    __bf16* d = &sB[buf][r * LDP + cc];
                    d[0] = (__bf16)f.x; d[1] = (__bf16)f.y; d[2] = (__bf16)f.z; d[3] = (__bf16)f.w;
                } else {                            // f32 [K,N]: transpose into LDS
                    int kk = pos >> NSH;            // k: 0..31
                    int nn = (pos & ((1 << NSH) - 1)) << 2;  // n
                    float4 f = *(const float4*)(Bf + (long long)(k0 + kk) * ldb + (n0 + nn));
                    sB[buf][(nn + 0) * LDP + kk] = (__bf16)f.x;
                    sB[buf][(nn + 1) * LDP + kk] = (__bf16)f.y;
                    sB[buf][(nn + 2) * LDP + kk] = (__bf16)f.z;
                    sB[buf][(nn + 3) * LDP + kk] = (__bf16)f.w;
                }
            }
        }
    };
    auto asyncWait = [&]() {
        if (BSRC == 2)
            asm volatile("s_wait_asynccnt 0x0" ::: "memory");
    };
    auto compute = [&](int buf) {
        V16U a[MF];
        #pragma unroll
        for (int m = 0; m < MF; ++m) {
            const __bf16* p = &sA[buf][(wm + 16 * m + l16) * LDP + (hi << 3)];
            a[m].h[0] = *(const bfx8*)(p);
            a[m].h[1] = *(const bfx8*)(p + 16);
        }
        V16U b[NF];
        #pragma unroll
        for (int n = 0; n < NF; ++n) {
            const __bf16* p = &sB[buf][(wn + 16 * n + l16) * LDP + (hi << 3)];
            b[n].h[0] = *(const bfx8*)(p);
            b[n].h[1] = *(const bfx8*)(p + 16);
        }
        #pragma unroll
        for (int m = 0; m < MF; ++m)
            #pragma unroll
            for (int n = 0; n < NF; ++n)
                acc[m][n] = __builtin_amdgcn_wmma_f32_16x16x32_bf16(
                    false, a[m].v, false, b[n].v, (short)0, acc[m][n], false, false);
    };

    // ---- main loop: double-buffered, one barrier per K-step ----
    stageA(0, 0);
    stageB(0, 0);
    asyncWait();
    __syncthreads();
    int buf = 0;
    for (int k0 = 0; k0 < K; k0 += BKK) {
        int kn = k0 + BKK;
        if (kn < K) {
            stageA(buf ^ 1, kn);
            stageB(buf ^ 1, kn);
        }
        // prefetch tiles two steps ahead into the cache hierarchy
        int kp = k0 + 2 * BKK;
        if (kp < K) {
            __builtin_prefetch((const void*)(A + (long long)(m0 + (tid & (BMt - 1))) * lda + kp), 0, 1);
            if (BSRC == 1)
                __builtin_prefetch((const void*)(Bf + (long long)(n0 + (tid & (BNt - 1))) * ldb + kp), 0, 1);
            else if (BSRC == 0)
                __builtin_prefetch((const void*)(Bf + (long long)(kp + (tid & 31)) * ldb + n0), 0, 1);
        }
        compute(buf);
        asyncWait();
        __syncthreads();
        buf ^= 1;
    }

    // ---- epilogue: acc[m][n][e] -> (row = e + 8*hi, col = l16) ----
    #pragma unroll
    for (int m = 0; m < MF; ++m) {
        const int mB = m0 + wm + 16 * m + (hi << 3);
        #pragma unroll
        for (int n = 0; n < NF; ++n) {
            const int nA = n0 + wn + 16 * n + l16;
            #pragma unroll
            for (int e = 0; e < 8; ++e) {
                long long r = mB + e;
                float v = acc[m][n][e];
                if (bias) v += bias[nA];
                if (res)  v += res[r * ldc + nA];
                C[r * ldc + nA] = v;
            }
        }
    }
}

// ---------------------------------------------------------------------------
// Tiled transpose + f32 -> bf16 convert:  src[R][C] f32  ->  dst[C][R] bf16
// ---------------------------------------------------------------------------
__global__ void k_transpose_bf16(const float* __restrict__ src, __bf16* __restrict__ dst,
                                 int R, int C)
{
    __shared__ __bf16 t[32][33];
    const int c0 = blockIdx.x * 32, r0 = blockIdx.y * 32;
    const int tx = threadIdx.x & 31, ty = threadIdx.x >> 5;   // 32 x 8
    #pragma unroll
    for (int j = 0; j < 32; j += 8)
        t[ty + j][tx] = (__bf16)src[(long long)(r0 + ty + j) * C + (c0 + tx)];
    __syncthreads();
    #pragma unroll
    for (int j = 0; j < 32; j += 8)
        dst[(long long)(c0 + ty + j) * R + (r0 + tx)] = t[tx][ty + j];
}

// ---------------------------------------------------------------------------
// Block reductions (256 threads = 8 wave32 waves)
// ---------------------------------------------------------------------------
__device__ __forceinline__ float blk_reduce_sum(float v) {
    #pragma unroll
    for (int o = 16; o > 0; o >>= 1) v += __shfl_xor(v, o, 32);
    __shared__ float sm[8];
    int lane = threadIdx.x & 31, wid = threadIdx.x >> 5;
    if (lane == 0) sm[wid] = v;
    __syncthreads();
    float r = sm[0];
    #pragma unroll
    for (int w = 1; w < 8; ++w) r += sm[w];
    __syncthreads();
    return r;
}

__device__ __forceinline__ float blk_reduce_max(float v) {
    #pragma unroll
    for (int o = 16; o > 0; o >>= 1) v = fmaxf(v, __shfl_xor(v, o, 32));
    __shared__ float sm[8];
    int lane = threadIdx.x & 31, wid = threadIdx.x >> 5;
    if (lane == 0) sm[wid] = v;
    __syncthreads();
    float r = sm[0];
    #pragma unroll
    for (int w = 1; w < 8; ++w) r = fmaxf(r, sm[w]);
    __syncthreads();
    return r;
}

// ---------------------------------------------------------------------------
// Elementwise / normalization kernels
// ---------------------------------------------------------------------------
__global__ void k_rmsnorm(const float* __restrict__ x, const float* __restrict__ w,
                          float* __restrict__ y, int n)
{
    const float* row = x + (long long)blockIdx.x * n;
    float s = 0.f;
    for (int i = threadIdx.x; i < n; i += 256) { float t = row[i]; s += t * t; }
    s = blk_reduce_sum(s);
    float sc = rsqrtf(s / (float)n + RMSEPS);
    float* o = y + (long long)blockIdx.x * n;
    for (int i = threadIdx.x; i < n; i += 256) o[i] = w[i] * row[i] * sc;
}

__global__ void k_rope(float* __restrict__ x, int T, int H, int hd, int npair)
{
    int idx = blockIdx.x * 256 + threadIdx.x;
    if (idx >= npair) return;
    int hp = hd >> 1;
    int p  = idx % hp;
    int rh = idx / hp;           // (b*T + t)*H + h
    int bt = rh / H;
    int t  = bt % T;
    long long base = (long long)rh * hd;
    float inv = __expf(-((float)(2 * p) / (float)hd) * 9.21034037198f); // ln(1e4)
    float ang = (float)t * inv;
    float c = __cosf(ang), s = __sinf(ang);
    float xr = x[base + 2 * p], xi = x[base + 2 * p + 1];
    x[base + 2 * p]     = xr * c - xi * s;
    x[base + 2 * p + 1] = xr * s + xi * c;
}

__global__ void k_softmax(float* __restrict__ x, int n, float scale)
{
    float* row = x + (long long)blockIdx.x * n;
    float mx = -3.4e38f;
    for (int i = threadIdx.x; i < n; i += 256) mx = fmaxf(mx, row[i] * scale);
    mx = blk_reduce_max(mx);
    float s = 0.f;
    for (int i = threadIdx.x; i < n; i += 256) {
        float e = __expf(row[i] * scale - mx);
        row[i] = e; s += e;
    }
    s = blk_reduce_sum(s);
    float inv = 1.f / s;
    for (int i = threadIdx.x; i < n; i += 256) row[i] *= inv;
}

__global__ void k_silu_mul(float* __restrict__ g, const float* __restrict__ u, long long n)
{
    long long i = (long long)blockIdx.x * 256 + threadIdx.x;
    if (i < n) { float a = g[i]; g[i] = (a / (1.f + __expf(-a))) * u[i]; }
}

__global__ void k_fuse(const float* __restrict__ cur, const float* __restrict__ emb,
                       const int* __restrict__ tgt, int ci, float* __restrict__ out,
                       long long ntot)
{
    long long idx = (long long)blockIdx.x * 256 + threadIdx.x;
    if (idx >= ntot) return;
    int r = (int)(idx >> 11);        // / 2048
    int c = (int)(idx & 2047);
    float v;
    if (c < HIDD) {
        v = cur[(long long)r * HIDD + c];
    } else {
        int t = tgt[r * NCODE + ci] + ci * VOC;
        v = emb[(long long)t * HIDD + (c - HIDD)];
    }
    out[idx] = v;
}

// ---------------------------------------------------------------------------
// Host-side orchestration
// ---------------------------------------------------------------------------
struct LayerP { const float *attn_norm,*ffn_norm,*w1,*w2,*w3,*wk,*wo,*wq,*wv; };
struct DecP   { LayerP L[6]; int nl; const float* norm; };

struct BatchCfg {
    int nz = 1, bH = 1, nrep = 1;
    long long sAo = 0, sAi = 0, sBo = 0, sBi = 0, sCo = 0, sCi = 0;
};

// bsrc: 0 = f32 [K,N], 1 = f32 [N,K] (B^T), 2 = bf16 [N,K] (async path)
static inline void launch_gemm(hipStream_t st,
    const float* A, long long lda, const float* B, long long ldb,
    float* C, long long ldc, const float* bias, const float* res,
    int M, int N, int K, int bsrc, const BatchCfg& bc)
{
    dim3 blk(256);
    if (bsrc == 1) {
        dim3 grid(N / 128, M / 128, bc.nz);
        gemm_bf16_wmma<1, 2, 4><<<grid, blk, 0, st>>>(A, lda, B, ldb, C, ldc, bias, res,
            M, N, K, bc.bH, bc.nrep, bc.sAo, bc.sAi, bc.sBo, bc.sBi, bc.sCo, bc.sCi);
    } else if (bsrc == 2) {
        dim3 grid(N / 128, M / 128, bc.nz);
        gemm_bf16_wmma<2, 2, 4><<<grid, blk, 0, st>>>(A, lda, B, ldb, C, ldc, bias, res,
            M, N, K, bc.bH, bc.nrep, bc.sAo, bc.sAi, bc.sBo, bc.sBi, bc.sCo, bc.sCi);
    } else if ((N % 128) == 0 && (M % 128) == 0) {
        dim3 grid(N / 128, M / 128, bc.nz);
        gemm_bf16_wmma<0, 2, 4><<<grid, blk, 0, st>>>(A, lda, B, ldb, C, ldc, bias, res,
            M, N, K, bc.bH, bc.nrep, bc.sAo, bc.sAi, bc.sBo, bc.sBi, bc.sCo, bc.sCi);
    } else {
        dim3 grid(N / 64, M / 64, bc.nz);
        gemm_bf16_wmma<0, 1, 2><<<grid, blk, 0, st>>>(A, lda, B, ldb, C, ldc, bias, res,
            M, N, K, bc.bH, bc.nrep, bc.sAo, bc.sAi, bc.sBo, bc.sBi, bc.sCo, bc.sCi);
    }
}

static void run_decoder_host(hipStream_t st, float* h, const DecP& dp, int H, int nkv,
                             float* xn, float* q, float* kb, float* vb,
                             float* sc, float* ao, float* g, float* u,
                             __bf16* w1T, __bf16* w3T, __bf16* w2T, float* outn)
{
    const int D = HIDD;
    const int hd  = D / H;
    const int kvd = nkv * hd;
    const int nrep = H / nkv;
    BatchCfg nb;   // non-batched defaults
    for (int l = 0; l < dp.nl; ++l) {
        const LayerP& L = dp.L[l];
        // --- attention ---
        k_rmsnorm<<<MTOK, 256, 0, st>>>(h, L.attn_norm, xn, D);
        launch_gemm(st, xn, D, L.wq, D,   q,  D,   nullptr, nullptr, MTOK, D,   D, 0, nb);
        launch_gemm(st, xn, D, L.wk, kvd, kb, kvd, nullptr, nullptr, MTOK, kvd, D, 0, nb);
        launch_gemm(st, xn, D, L.wv, kvd, vb, kvd, nullptr, nullptr, MTOK, kvd, D, 0, nb);
        int pq = MTOK * H * hd / 2;
        k_rope<<<(pq + 255) / 256, 256, 0, st>>>(q,  TT, H,   hd, pq);
        int pk = MTOK * nkv * hd / 2;
        k_rope<<<(pk + 255) / 256, 256, 0, st>>>(kb, TT, nkv, hd, pk);

        BatchCfg bs; bs.nz = BB * H; bs.bH = H; bs.nrep = nrep;
        bs.sAo = (long long)TT * D;       bs.sAi = hd;
        bs.sBo = (long long)TT * kvd;     bs.sBi = hd;
        bs.sCo = (long long)H * TT * TT;  bs.sCi = (long long)TT * TT;
        // scores[b,h] = Q * K^T  (M=T, N=T, K=hd)
        launch_gemm(st, q, D, kb, kvd, sc, TT, nullptr, nullptr, TT, TT, hd, 1, bs);
        k_softmax<<<BB * H * TT, 256, 0, st>>>(sc, TT, 1.0f / sqrtf((float)hd));

        BatchCfg bo; bo.nz = BB * H; bo.bH = H; bo.nrep = nrep;
        bo.sAo = (long long)H * TT * TT;  bo.sAi = (long long)TT * TT;
        bo.sBo = (long long)TT * kvd;     bo.sBi = hd;
        bo.sCo = (long long)TT * D;       bo.sCi = hd;
        // out[b,h] = probs * V  (M=T, N=hd, K=T)
        launch_gemm(st, sc, TT, vb, kvd, ao, D, nullptr, nullptr, TT, hd, TT, 0, bo);
        // h += out * wo
        launch_gemm(st, ao, D, L.wo, D, h, D, nullptr, h, MTOK, D, D, 0, nb);

        // --- FFN (SwiGLU) --- pre-transpose weights to bf16 [N,K] for the
        // async-to-LDS GEMM path (w1/w3: [D,FFD] -> [FFD,D]; w2: [FFD,D] -> [D,FFD])
        k_transpose_bf16<<<dim3(FFD / 32, HIDD / 32), 256, 0, st>>>(L.w1, w1T, HIDD, FFD);
        k_transpose_bf16<<<dim3(FFD / 32, HIDD / 32), 256, 0, st>>>(L.w3, w3T, HIDD, FFD);
        k_transpose_bf16<<<dim3(HIDD / 32, FFD / 32), 256, 0, st>>>(L.w2, w2T, FFD, HIDD);

        k_rmsnorm<<<MTOK, 256, 0, st>>>(h, L.ffn_norm, xn, D);
        launch_gemm(st, xn, D, (const float*)w1T, D, g, FFD, nullptr, nullptr,
                    MTOK, FFD, D, 2, nb);
        launch_gemm(st, xn, D, (const float*)w3T, D, u, FFD, nullptr, nullptr,
                    MTOK, FFD, D, 2, nb);
        long long nsu = (long long)MTOK * FFD;
        k_silu_mul<<<(int)((nsu + 255) / 256), 256, 0, st>>>(g, u, nsu);
        launch_gemm(st, g, FFD, (const float*)w2T, FFD, h, D, nullptr, h,
                    MTOK, D, FFD, 2, nb);
    }
    k_rmsnorm<<<MTOK, 256, 0, st>>>(h, dp.norm, outn, HIDD);
}

// ---------------------------------------------------------------------------
// Entry point
// ---------------------------------------------------------------------------
extern "C" void kernel_launch(void* const* d_in, const int* in_sizes, int n_in,
                              void* d_out, int out_size, void* d_ws, size_t ws_size,
                              hipStream_t stream)
{
    (void)in_sizes; (void)out_size; (void)ws_size;
    if (!d_in || !d_out || !d_ws) return;
    if (n_in < 220) return;   // expected JAX pytree leaf count

    // ---- parse inputs in pytree order (sorted dict keys, lists in order) ----
    int c = 0;
    auto nf = [&]() -> const float* { return (const float*)d_in[c++]; };
    auto read_layer = [&](LayerP& lp) {
        lp.attn_norm = nf(); lp.ffn_norm = nf();
        lp.w1 = nf(); lp.w2 = nf(); lp.w3 = nf();
        lp.wk = nf(); lp.wo = nf(); lp.wq = nf(); lp.wv = nf();
    };
    auto read_dec = [&](DecP& d, int nl) {
        d.nl = nl;
        for (int i = 0; i < nl; ++i) read_layer(d.L[i]);
        d.norm = nf();
    };

    const float* audio_emb = nf();                 // params.audio_emb
    DecP coarse;  read_dec(coarse, 6);             // params.coarse
    const float* ch_b = nf();                      // params.coarse_head.b
    const float* ch_w = nf();                      // params.coarse_head.w
    const float *fh_b[NCODE - 1], *fh_w[NCODE - 1];
    for (int i = 0; i < NCODE - 1; ++i) { fh_b[i] = nf(); fh_w[i] = nf(); }  // fine_heads
    const float *fu_b[NCODE - 1], *fu_w[NCODE - 1];
    for (int i = 0; i < NCODE - 1; ++i) { fu_b[i] = nf(); fu_w[i] = nf(); }  // fusion
    const float* ip_b = nf();                      // params.input_proj.b
    const float* ip_w = nf();                      // params.input_proj.w
    DecP pri[3];  for (int i = 0; i < 3; ++i) read_dec(pri[i], 4);           // priority
    DecP shr;     read_dec(shr, 2);                // shared
    const int*   targets = (const int*)d_in[c++];  // targets
    const float* text    = (const float*)d_in[c++];// text_features

    // ---- workspace bump allocator ----
    char* wp = (char*)d_ws;
    auto wsa = [&](size_t bytes) -> void* {
        void* p = (void*)wp;
        wp += (bytes + 255) & ~(size_t)255;
        return p;
    };
    float* h    = (float*)wsa((size_t)MTOK * HIDD * 4);
    float* xn   = (float*)wsa((size_t)MTOK * HIDD * 4);
    float* q    = (float*)wsa((size_t)MTOK * HIDD * 4);
    float* kb   = (float*)wsa((size_t)MTOK * 256  * 4);
    float* vb   = (float*)wsa((size_t)MTOK * 256  * 4);
    float* sc   = (float*)wsa((size_t)BB * 16 * TT * TT * 4);  // worst case: 16 heads
    float* ao   = (float*)wsa((size_t)MTOK * HIDD * 4);
    float* g    = (float*)wsa((size_t)MTOK * FFD  * 4);
    float* u    = (float*)wsa((size_t)MTOK * FFD  * 4);
    float* cur  = (float*)wsa((size_t)MTOK * HIDD * 4);
    float* cat  = (float*)wsa((size_t)MTOK * 2 * HIDD * 4);
    __bf16* w1T = (__bf16*)wsa((size_t)FFD * HIDD * 2);
    __bf16* w3T = (__bf16*)wsa((size_t)FFD * HIDD * 2);
    __bf16* w2T = (__bf16*)wsa((size_t)FFD * HIDD * 2);

    float* outp = (float*)d_out;
    const long long ldcOut = (long long)NCODE * VOC;
    BatchCfg nb;

    // ---- 1) input projection: h = text @ Wi + bi ----
    launch_gemm(stream, text, 512, ip_w, HIDD, h, HIDD, ip_b, nullptr,
                MTOK, HIDD, 512, 0, nb);

    // ---- 2) coarse decoder (16 heads, 2 kv) -> cur ----
    run_decoder_host(stream, h, coarse, 16, 2, xn, q, kb, vb, sc, ao, g, u,
                     w1T, w3T, w2T, cur);

    // ---- 3) coarse head -> logits[:, :, 0, :] ----
    launch_gemm(stream, cur, HIDD, ch_w, VOC, outp, ldcOut, ch_b, nullptr,
                MTOK, VOC, HIDD, 0, nb);

    // ---- 4) fine stages ----
    for (int i = 0; i < NCODE - 1; ++i) {
        long long ntot = (long long)MTOK * 2 * HIDD;
        k_fuse<<<(int)((ntot + 255) / 256), 256, 0, stream>>>(cur, audio_emb, targets,
                                                              i, cat, ntot);
        launch_gemm(stream, cat, 2 * HIDD, fu_w[i], HIDD, h, HIDD, fu_b[i], nullptr,
                    MTOK, HIDD, 2 * HIDD, 0, nb);
        const DecP& dp = (i < 3) ? pri[i] : shr;
        run_decoder_host(stream, h, dp, 8, 2, xn, q, kb, vb, sc, ao, g, u,
                         w1T, w3T, w2T, cur);
        launch_gemm(stream, cur, HIDD, fh_w[i], VOC, outp + (long long)(i + 1) * VOC,
                    ldcOut, fh_b[i], nullptr, MTOK, VOC, HIDD, 0, nb);
    }
}
